// LatticeSelfAttention_46488726012521
// MI455X (gfx1250) — matrix operand order repaired and checked by reading, not compiled, not run on hardware
//
#include <hip/hip_runtime.h>
#include <hip/hip_bf16.h>

#define B_  8
#define S_  2048
#define H_  512
#define NH_ 8
#define HD_ 64
#define L_  4096
#define M_  (B_ * S_)   // 16384 rows

typedef __attribute__((ext_vector_type(16))) _Float16 v16h;
typedef __attribute__((ext_vector_type(8)))  _Float16 v8h;
typedef __attribute__((ext_vector_type(4)))  _Float16 v4h;
typedef __attribute__((ext_vector_type(8)))  float    v8f;

// ---------------------------------------------------------------------------
// CDNA5 async global->LDS copy (GLOBAL_LOAD_ASYNC_TO_LDS_B128, ASYNCcnt).
// saddr form: vdst = per-lane LDS byte offset, vaddr = per-lane byte offset,
// saddr = common 64-bit base.  Each lane moves 16 bytes.
// ---------------------------------------------------------------------------
static __device__ inline void async_ld_b128(void* lds_dst, const void* gbase,
                                            unsigned voff_bytes) {
  asm volatile("global_load_async_to_lds_b128 %0, %1, %2"
               :: "v"((unsigned)(size_t)lds_dst),            // LDS offset (low 32b)
                  "v"(voff_bytes),
                  "s"((unsigned long long)(size_t)gbase)
               : "memory");
}

static __device__ inline void wait_async0() {
#if __has_builtin(__builtin_amdgcn_s_wait_asynccnt)
  __builtin_amdgcn_s_wait_asynccnt(0);
#else
  asm volatile("s_wait_asynccnt 0x0" ::: "memory");
#endif
}

// ---------------------------------------------------------------------------
// WMMA fragment loaders per CDNA5 ISA 7.12.2 (wave32).
// Both loaders read two contiguous 16-byte runs per lane -> ds_load_b128.
// ---------------------------------------------------------------------------
static __device__ inline v16h load_A_frag(const _Float16* p, int ld) {
  int lane = threadIdx.x & 31;
  int row  = lane & 15;
  int kb   = (lane & 16) ? 8 : 0;
  v16h a;
#pragma unroll
  for (int j = 0; j < 8; ++j) a[j] = p[row * ld + kb + j];
#pragma unroll
  for (int j = 0; j < 8; ++j) a[8 + j] = p[row * ld + kb + 16 + j];
  return a;
}

// B operand stored [N][K] row-major (we always keep B transposed in LDS)
static __device__ inline v16h load_B_fragT(const _Float16* p, int ld) {
  int lane = threadIdx.x & 31;
  int col  = lane & 15;
  int kb   = (lane & 16) ? 8 : 0;
  v16h b;
#pragma unroll
  for (int j = 0; j < 8; ++j) b[j] = p[col * ld + kb + j];
#pragma unroll
  for (int j = 0; j < 8; ++j) b[8 + j] = p[col * ld + kb + 16 + j];
  return b;
}

// ---------------------------------------------------------------------------
__global__ __launch_bounds__(256) void cvt_f32_f16(const float* __restrict__ src,
                                                   _Float16* __restrict__ dst, int n4) {
  int i = blockIdx.x * blockDim.x + threadIdx.x;
  if (i < n4) {
    float4 v = *(const float4*)(src + (size_t)i * 4);
    v4h h;
    h[0] = (_Float16)v.x; h[1] = (_Float16)v.y;
    h[2] = (_Float16)v.z; h[3] = (_Float16)v.w;
    *(v4h*)(dst + (size_t)i * 4) = h;
  }
}

// Weight convert + transpose: dst[n*H + k] = (f16) src[k*H + n]
__global__ __launch_bounds__(256) void cvt_transpose_f32_f16(const float* __restrict__ src,
                                                             _Float16* __restrict__ dst) {
  __shared__ float tile[32][33];
  int n0 = blockIdx.x * 32, k0 = blockIdx.y * 32;
  int tx = threadIdx.x & 31, ty = threadIdx.x >> 5;
#pragma unroll
  for (int i = 0; i < 32; i += 8)
    tile[ty + i][tx] = src[(size_t)(k0 + ty + i) * H_ + n0 + tx];
  __syncthreads();
#pragma unroll
  for (int i = 0; i < 32; i += 8)
    dst[(size_t)(n0 + ty + i) * H_ + k0 + tx] = (_Float16)tile[tx][ty + i];
}

// bias[b][s][t] = (mask[b][t]==0) ? -1e9 : 0   then lexicon scatter (+1.5)
__global__ __launch_bounds__(256) void bias_init(const int* __restrict__ amask,
                                                 float* __restrict__ bias) {
  size_t i4  = ((size_t)blockIdx.x * blockDim.x + threadIdx.x) * 4;
  size_t b   = i4 / ((size_t)S_ * S_);
  size_t rem = i4 % ((size_t)S_ * S_);
  int t = (int)(rem & (S_ - 1));
  const int* mb = amask + b * S_;
  float4 v;
  v.x = mb[t + 0] ? 0.f : -1e9f;
  v.y = mb[t + 1] ? 0.f : -1e9f;
  v.z = mb[t + 2] ? 0.f : -1e9f;
  v.w = mb[t + 3] ? 0.f : -1e9f;
  *(float4*)(bias + i4) = v;
}

__global__ __launch_bounds__(256) void bias_scatter(const int* __restrict__ ch,
                                                    const int* __restrict__ wd,
                                                    float* __restrict__ bias) {
  int i = blockIdx.x * blockDim.x + threadIdx.x;   // B*L threads
  int b = i >> 12;                                 // L = 4096
  atomicAdd(bias + ((size_t)b * S_ + ch[i]) * S_ + wd[i], 1.5f);
}

// ---------------------------------------------------------------------------
// Projection GEMM with async double-buffered staging.
// Y = X(f16)[M,512] * W + bias, W supplied TRANSPOSED (Wt[n][k]) so every LDS
// fragment read is a contiguous b128 pair.
// ---------------------------------------------------------------------------
template <bool TRANSPOSE_OUT>
__global__ __launch_bounds__(128) void proj_gemm(
    const _Float16* __restrict__ X, const _Float16* __restrict__ Wt,
    const float* __restrict__ bias, _Float16* __restrict__ out) {
  __shared__ __align__(16) _Float16 As[2][64 * 32];   // [m][k]
  __shared__ __align__(16) _Float16 Bs[2][64 * 32];   // [n][k]
  int m0 = blockIdx.x * 64, n0 = blockIdx.y * 64;
  int wid = threadIdx.x >> 5, lane = threadIdx.x & 31;

  // per-thread staging geometry: 2 x 16B chunks per tile
  int row0 = (threadIdx.x * 2) >> 2,      coff0 = ((threadIdx.x * 2) & 3) << 3;
  int row1 = (threadIdx.x * 2 + 1) >> 2,  coff1 = ((threadIdx.x * 2 + 1) & 3) << 3;

  v8f acc[4];
#pragma unroll
  for (int n = 0; n < 4; ++n)
#pragma unroll
    for (int r = 0; r < 8; ++r) acc[n][r] = 0.f;

  // prologue: stage k-tile 0 into buffer 0
  async_ld_b128(As[0] + row0 * 32 + coff0, X, (unsigned)(((size_t)(m0 + row0) * H_ + coff0) * 2));
  async_ld_b128(As[0] + row1 * 32 + coff1, X, (unsigned)(((size_t)(m0 + row1) * H_ + coff1) * 2));
  async_ld_b128(Bs[0] + row0 * 32 + coff0, Wt, (unsigned)(((size_t)(n0 + row0) * H_ + coff0) * 2));
  async_ld_b128(Bs[0] + row1 * 32 + coff1, Wt, (unsigned)(((size_t)(n0 + row1) * H_ + coff1) * 2));
  wait_async0();
  __syncthreads();

  for (int kk = 0; kk < H_ / 32; ++kk) {
    int cur = kk & 1, nxt = cur ^ 1;
    if (kk + 1 < H_ / 32) {           // stage next k-tile while computing
      int k1 = (kk + 1) * 32;
      async_ld_b128(As[nxt] + row0 * 32 + coff0, X, (unsigned)(((size_t)(m0 + row0) * H_ + k1 + coff0) * 2));
      async_ld_b128(As[nxt] + row1 * 32 + coff1, X, (unsigned)(((size_t)(m0 + row1) * H_ + k1 + coff1) * 2));
      async_ld_b128(Bs[nxt] + row0 * 32 + coff0, Wt, (unsigned)(((size_t)(n0 + row0) * H_ + k1 + coff0) * 2));
      async_ld_b128(Bs[nxt] + row1 * 32 + coff1, Wt, (unsigned)(((size_t)(n0 + row1) * H_ + k1 + coff1) * 2));
    }
    v16h a = load_A_frag(As[cur] + wid * 16 * 32, 32);
#pragma unroll
    for (int n = 0; n < 4; ++n) {
      v16h bf = load_B_fragT(Bs[cur] + n * 16 * 32, 32);
      acc[n] = __builtin_amdgcn_wmma_f32_16x16x32_f16(false, a, false, bf,
                                                      (short)0, acc[n], false, false);
    }
    wait_async0();
    __syncthreads();
  }

  int roff = (lane & 16) ? 8 : 0;
#pragma unroll
  for (int n = 0; n < 4; ++n) {
    int col = n0 + n * 16 + (lane & 15);
    int h = col >> 6, d = col & 63;
    float bv = bias[col];
#pragma unroll
    for (int r = 0; r < 8; ++r) {
      int m = m0 + wid * 16 + r + roff;
      int b = m >> 11, s = m & 2047;
      float val = acc[n][r] + bv;
      if (TRANSPOSE_OUT)
        out[(((size_t)(b * NH_ + h)) * HD_ + d) * S_ + s] = (_Float16)val;
      else
        out[(((size_t)(b * NH_ + h)) * S_ + s) * HD_ + d] = (_Float16)val;
    }
  }
}

// Output GEMM: out(f32)[M,512] = ctx(f16)[M,512] * Wo (transposed) + bo
__global__ __launch_bounds__(128) void out_gemm(
    const _Float16* __restrict__ X, const _Float16* __restrict__ Wt,
    const float* __restrict__ bias, float* __restrict__ out) {
  __shared__ __align__(16) _Float16 As[2][64 * 32];
  __shared__ __align__(16) _Float16 Bs[2][64 * 32];
  int m0 = blockIdx.x * 64, n0 = blockIdx.y * 64;
  int wid = threadIdx.x >> 5, lane = threadIdx.x & 31;

  int row0 = (threadIdx.x * 2) >> 2,      coff0 = ((threadIdx.x * 2) & 3) << 3;
  int row1 = (threadIdx.x * 2 + 1) >> 2,  coff1 = ((threadIdx.x * 2 + 1) & 3) << 3;

  v8f acc[4];
#pragma unroll
  for (int n = 0; n < 4; ++n)
#pragma unroll
    for (int r = 0; r < 8; ++r) acc[n][r] = 0.f;

  async_ld_b128(As[0] + row0 * 32 + coff0, X, (unsigned)(((size_t)(m0 + row0) * H_ + coff0) * 2));
  async_ld_b128(As[0] + row1 * 32 + coff1, X, (unsigned)(((size_t)(m0 + row1) * H_ + coff1) * 2));
  async_ld_b128(Bs[0] + row0 * 32 + coff0, Wt, (unsigned)(((size_t)(n0 + row0) * H_ + coff0) * 2));
  async_ld_b128(Bs[0] + row1 * 32 + coff1, Wt, (unsigned)(((size_t)(n0 + row1) * H_ + coff1) * 2));
  wait_async0();
  __syncthreads();

  for (int kk = 0; kk < H_ / 32; ++kk) {
    int cur = kk & 1, nxt = cur ^ 1;
    if (kk + 1 < H_ / 32) {
      int k1 = (kk + 1) * 32;
      async_ld_b128(As[nxt] + row0 * 32 + coff0, X, (unsigned)(((size_t)(m0 + row0) * H_ + k1 + coff0) * 2));
      async_ld_b128(As[nxt] + row1 * 32 + coff1, X, (unsigned)(((size_t)(m0 + row1) * H_ + k1 + coff1) * 2));
      async_ld_b128(Bs[nxt] + row0 * 32 + coff0, Wt, (unsigned)(((size_t)(n0 + row0) * H_ + k1 + coff0) * 2));
      async_ld_b128(Bs[nxt] + row1 * 32 + coff1, Wt, (unsigned)(((size_t)(n0 + row1) * H_ + k1 + coff1) * 2));
    }
    v16h a = load_A_frag(As[cur] + wid * 16 * 32, 32);
#pragma unroll
    for (int n = 0; n < 4; ++n) {
      v16h bf = load_B_fragT(Bs[cur] + n * 16 * 32, 32);
      acc[n] = __builtin_amdgcn_wmma_f32_16x16x32_f16(false, a, false, bf,
                                                      (short)0, acc[n], false, false);
    }
    wait_async0();
    __syncthreads();
  }

  int roff = (lane & 16) ? 8 : 0;
#pragma unroll
  for (int n = 0; n < 4; ++n) {
    int col = n0 + n * 16 + (lane & 15);
    float bv = bias[col];
#pragma unroll
    for (int r = 0; r < 8; ++r) {
      int m = m0 + wid * 16 + r + roff;
      out[(size_t)m * H_ + col] = acc[n][r] + bv;
    }
  }
}

// ---------------------------------------------------------------------------
// Flash attention, async double-buffered K/V staging.
// ---------------------------------------------------------------------------
__global__ __launch_bounds__(128) void flash_attn(
    const _Float16* __restrict__ Qb, const _Float16* __restrict__ Kb,
    const _Float16* __restrict__ Vtb, const float* __restrict__ bias,
    _Float16* __restrict__ ctx) {
  __shared__ __align__(16) _Float16 Qs[64 * 64];        // [qrow][d]
  __shared__ __align__(16) _Float16 Ks[2][64 * 64];     // [key][d]
  __shared__ __align__(16) _Float16 Vs[2][64 * 64];     // [d][key]  (transposed V)
  __shared__ __align__(16) _Float16 Ps[64 * 64];        // [qrow][key]

  int qt = blockIdx.x, h = blockIdx.y, b = blockIdx.z;
  int wid = threadIdx.x >> 5, lane = threadIdx.x & 31;
  const _Float16* Q  = Qb  + ((size_t)(b * NH_ + h)) * S_ * HD_;
  const _Float16* K  = Kb  + ((size_t)(b * NH_ + h)) * S_ * HD_;
  const _Float16* Vt = Vtb + ((size_t)(b * NH_ + h)) * HD_ * S_;
  const float* bias_b = bias + (size_t)b * S_ * S_;

  int vrow[4], vcoff[4];
#pragma unroll
  for (int i = 0; i < 4; ++i) {
    int c = threadIdx.x + i * 128;
    vrow[i] = c >> 3; vcoff[i] = (c & 7) << 3;
  }

  // prologue: async-stage Q tile and K/V tile 0 into buffer 0
#pragma unroll
  for (int i = 0; i < 4; ++i) {
    int c = threadIdx.x + i * 128;
    async_ld_b128(Qs + c * 8, Q, (unsigned)(((size_t)qt * 4096 + c * 8) * 2));
    async_ld_b128(Ks[0] + c * 8, K, (unsigned)((size_t)c * 16));
    async_ld_b128(Vs[0] + vrow[i] * 64 + vcoff[i], Vt,
                  (unsigned)(((size_t)vrow[i] * S_ + vcoff[i]) * 2));
  }
  wait_async0();
  __syncthreads();

  v16h aq0 = load_A_frag(Qs + wid * 16 * 64, 64);
  v16h aq1 = load_A_frag(Qs + wid * 16 * 64 + 32, 64);

  float mrow[8], lrow[8];
  v8f accO[4];
#pragma unroll
  for (int r = 0; r < 8; ++r) { mrow[r] = -1e30f; lrow[r] = 0.f; }
#pragma unroll
  for (int n = 0; n < 4; ++n)
#pragma unroll
    for (int r = 0; r < 8; ++r) accO[n][r] = 0.f;

  int roff  = (lane & 16) ? 8 : 0;
  int qrow0 = qt * 64 + wid * 16 + roff;

  for (int kt = 0; kt < S_ / 64; ++kt) {
    int cur = kt & 1, nxt = cur ^ 1;
    if (kt + 1 < S_ / 64) {           // stage next K/V tile during compute
#pragma unroll
      for (int i = 0; i < 4; ++i) {
        int c = threadIdx.x + i * 128;
        async_ld_b128(Ks[nxt] + c * 8, K,
                      (unsigned)(((size_t)(kt + 1) * 4096 + c * 8) * 2));
        async_ld_b128(Vs[nxt] + vrow[i] * 64 + vcoff[i], Vt,
                      (unsigned)(((size_t)vrow[i] * S_ + (size_t)(kt + 1) * 64 + vcoff[i]) * 2));
      }
    }

    // S = Q * K^T  (fp32 accum), 16x64 strip per wave
    v8f sc[4];
#pragma unroll
    for (int n = 0; n < 4; ++n) {
      v16h bk0 = load_B_fragT(Ks[cur] + n * 16 * 64, 64);
      v16h bk1 = load_B_fragT(Ks[cur] + n * 16 * 64 + 32, 64);
      v8f c;
#pragma unroll
      for (int r = 0; r < 8; ++r) c[r] = 0.f;
      c = __builtin_amdgcn_wmma_f32_16x16x32_f16(false, aq0, false, bk0, (short)0, c, false, false);
      c = __builtin_amdgcn_wmma_f32_16x16x32_f16(false, aq1, false, bk1, (short)0, c, false, false);
      sc[n] = c;
    }
    // scale (1/sqrt(64)) + mask/lexicon bias (fp32 domain)
#pragma unroll
    for (int n = 0; n < 4; ++n) {
      int scol = kt * 64 + n * 16 + (lane & 15);
#pragma unroll
      for (int r = 0; r < 8; ++r)
        sc[n][r] = sc[n][r] * 0.125f + bias_b[(size_t)(qrow0 + r) * S_ + scol];
    }
    // online softmax; row reductions stay inside each 16-lane half (xor<16)
    float pw[4][8];
#pragma unroll
    for (int r = 0; r < 8; ++r) {
      float rm = sc[0][r];
#pragma unroll
      for (int n = 1; n < 4; ++n) rm = fmaxf(rm, sc[n][r]);
#pragma unroll
      for (int s = 1; s < 16; s <<= 1) rm = fmaxf(rm, __shfl_xor(rm, s, 32));
      float nm   = fmaxf(mrow[r], rm);
      float corr = __expf(mrow[r] - nm);
      mrow[r] = nm;
      float rs = 0.f;
#pragma unroll
      for (int n = 0; n < 4; ++n) {
        float p = __expf(sc[n][r] - nm);
        pw[n][r] = p;
        rs += p;
      }
#pragma unroll
      for (int s = 1; s < 16; s <<= 1) rs += __shfl_xor(rs, s, 32);
      lrow[r] = lrow[r] * corr + rs;
#pragma unroll
      for (int n = 0; n < 4; ++n) accO[n][r] *= corr;
    }
    // transpose P strip through wave-private LDS into A-frag layout
#pragma unroll
    for (int n = 0; n < 4; ++n)
#pragma unroll
      for (int r = 0; r < 8; ++r)
        Ps[(size_t)(wid * 16 + r + roff) * 64 + n * 16 + (lane & 15)] = (_Float16)pw[n][r];

    v16h ap0 = load_A_frag(Ps + wid * 16 * 64, 64);
    v16h ap1 = load_A_frag(Ps + wid * 16 * 64 + 32, 64);
#pragma unroll
    for (int n = 0; n < 4; ++n) {
      v16h bv0 = load_B_fragT(Vs[cur] + n * 16 * 64, 64);   // V^T[d][key]
      v16h bv1 = load_B_fragT(Vs[cur] + n * 16 * 64 + 32, 64);
      accO[n] = __builtin_amdgcn_wmma_f32_16x16x32_f16(false, ap0, false, bv0, (short)0, accO[n], false, false);
      accO[n] = __builtin_amdgcn_wmma_f32_16x16x32_f16(false, ap1, false, bv1, (short)0, accO[n], false, false);
    }

    wait_async0();       // my async copies into buf^1 have landed
    __syncthreads();     // everyone done reading buf & done copying
  }

  // epilogue: O / l, write merged-head f16 ctx [B,S,H]
#pragma unroll
  for (int r = 0; r < 8; ++r) {
    float inv = 1.0f / fmaxf(lrow[r], 1e-30f);
    int srow  = qrow0 + r;
#pragma unroll
    for (int n = 0; n < 4; ++n) {
      int d = h * HD_ + n * 16 + (lane & 15);
      ctx[((size_t)(b * S_ + srow)) * H_ + d] = (_Float16)(accO[n][r] * inv);
    }
  }
}

// ---------------------------------------------------------------------------
extern "C" void kernel_launch(void* const* d_in, const int* in_sizes, int n_in,
                              void* d_out, int out_size, void* d_ws, size_t ws_size,
                              hipStream_t stream) {
  const float* hs  = (const float*)d_in[0];
  const float* Wq  = (const float*)d_in[1];
  const float* bq  = (const float*)d_in[2];
  const float* Wk  = (const float*)d_in[3];
  const float* bk  = (const float*)d_in[4];
  const float* Wv  = (const float*)d_in[5];
  const float* bv  = (const float*)d_in[6];
  const float* Wo  = (const float*)d_in[7];
  const float* bo  = (const float*)d_in[8];
  const int* amask = (const int*)d_in[9];
  const int* lch   = (const int*)d_in[10];
  const int* lwd   = (const int*)d_in[11];
  float* out = (float*)d_out;

  char* ws = (char*)d_ws;
  size_t off = 0;
  auto alloc = [&](size_t bytes) -> void* {
    void* p = ws + off;
    off += (bytes + 255) & ~(size_t)255;
    return p;
  };
  _Float16* hs16  = (_Float16*)alloc((size_t)M_ * H_ * 2);
  _Float16* wq16  = (_Float16*)alloc((size_t)H_ * H_ * 2);   // transposed [n][k]
  _Float16* wk16  = (_Float16*)alloc((size_t)H_ * H_ * 2);
  _Float16* wv16  = (_Float16*)alloc((size_t)H_ * H_ * 2);
  _Float16* wo16  = (_Float16*)alloc((size_t)H_ * H_ * 2);
  _Float16* q16   = (_Float16*)alloc((size_t)M_ * H_ * 2);   // [b,h,s,d]
  _Float16* k16   = (_Float16*)alloc((size_t)M_ * H_ * 2);   // [b,h,s,d]
  _Float16* vt16  = (_Float16*)alloc((size_t)M_ * H_ * 2);   // [b,h,d,s]
  _Float16* ctx16 = (_Float16*)alloc((size_t)M_ * H_ * 2);   // [b,s,H]
  float* biasM    = (float*)alloc((size_t)B_ * S_ * S_ * 4);

  // 1) precision converts (weights transposed to [n][k])
  cvt_f32_f16<<<(M_ * H_ / 4 + 255) / 256, 256, 0, stream>>>(hs, hs16, M_ * H_ / 4);
  dim3 tg(H_ / 32, H_ / 32);
  cvt_transpose_f32_f16<<<tg, 256, 0, stream>>>(Wq, wq16);
  cvt_transpose_f32_f16<<<tg, 256, 0, stream>>>(Wk, wk16);
  cvt_transpose_f32_f16<<<tg, 256, 0, stream>>>(Wv, wv16);
  cvt_transpose_f32_f16<<<tg, 256, 0, stream>>>(Wo, wo16);

  // 2) dense additive bias: mask(-1e9) then lexicon-link scatter (+1.5)
  bias_init<<<(size_t)B_ * S_ * S_ / 4 / 256, 256, 0, stream>>>(amask, biasM);
  bias_scatter<<<B_ * L_ / 256, 256, 0, stream>>>(lch, lwd, biasM);

  // 3) Q/K/V projections (WMMA, async staging);  V written transposed [b,h,d,s]
  dim3 gg(M_ / 64, H_ / 64);
  proj_gemm<false><<<gg, 128, 0, stream>>>(hs16, wq16, bq, q16);
  proj_gemm<false><<<gg, 128, 0, stream>>>(hs16, wk16, bk, k16);
  proj_gemm<true ><<<gg, 128, 0, stream>>>(hs16, wv16, bv, vt16);

  // 4) flash attention (async double-buffered K/V)
  flash_attn<<<dim3(S_ / 64, NH_, B_), 128, 0, stream>>>(q16, k16, vt16, biasM, ctx16);

  // 5) output projection -> fp32
  out_gemm<<<gg, 128, 0, stream>>>(ctx16, wo16, bo, out);
}
